// BipartiteGraph_8177617732032
// MI455X (gfx1250) — matrix operand order repaired and checked by reading, not compiled, not run on hardware
//
#include <hip/hip_runtime.h>

typedef __attribute__((ext_vector_type(16))) _Float16 v16h;
typedef __attribute__((ext_vector_type(8)))  _Float16 v8h;
typedef __attribute__((ext_vector_type(8)))  float    v8f;
typedef __attribute__((ext_vector_type(4)))  float    v4f;

#define NU   512
#define NV   512
#define KDIM 128
#define RDIM 256

__device__ __forceinline__ v16h join8(v8h lo, v8h hi) {
  v16h r;
#pragma unroll
  for (int i = 0; i < 8; ++i) { r[i] = lo[i]; r[i + 8] = hi[i]; }
  return r;
}

// ---------------------------------------------------------------- elementwise
__global__ void k_cvt_f16(const float* __restrict__ x, _Float16* __restrict__ y, int n) {
  int i = blockIdx.x * blockDim.x + threadIdx.x;
  if (i < n) y[i] = (_Float16)x[i];
}

__global__ void k_copy_f32(const float* __restrict__ x, float* __restrict__ y, int n) {
  int i = blockIdx.x * blockDim.x + threadIdx.x;
  if (i < n) y[i] = x[i];
}

// AT[c, r] = (f16) A[r, c]   (A is rows x cols)
__global__ void k_transpose_cvt(const float* __restrict__ A, _Float16* __restrict__ AT,
                                int rows, int cols) {
  int i = blockIdx.x * blockDim.x + threadIdx.x;
  if (i >= rows * cols) return;
  int r = i / cols, c = i - r * cols;
  AT[(size_t)c * rows + r] = (_Float16)A[i];
}

// y = (f16) concat(a[M,Na], b[M,Nb]) row-wise
__global__ void k_concat_cvt(const float* __restrict__ a, const float* __restrict__ b,
                             _Float16* __restrict__ y, int M, int Na, int Nb) {
  int i = blockIdx.x * blockDim.x + threadIdx.x;
  int Nt = Na + Nb;
  if (i >= M * Nt) return;
  int row = i / Nt, col = i - row * Nt;
  float v = (col < Na) ? a[(size_t)row * Na + col] : b[(size_t)row * Nb + (col - Na)];
  y[i] = (_Float16)v;
}

// ------------------------------------------------------------ generic WMMA GEMM
// out[M,N](f32) = opt_relu( X[M,Kd](f16,rm) @ W[Kd,N](f16,rm) + bias )
// one wave per 16x16 output tile; Kd multiple of 32.
__global__ void k_gemm_f16(const _Float16* __restrict__ X, const _Float16* __restrict__ W,
                           const float* __restrict__ bias, float* __restrict__ out,
                           int M, int Kd, int N, int relu) {
  int wave = (int)((blockIdx.x * blockDim.x + threadIdx.x) >> 5);
  int lane = threadIdx.x & 31;
  int ntiles = N >> 4;
  int mt = wave / ntiles;
  int nt = wave - mt * ntiles;
  if (mt >= (M >> 4)) return;
  int m = lane & 15;
  int half = lane >> 4;
  const _Float16* xrow = X + (size_t)(mt * 16 + m) * Kd;
  v8f acc = {};
  for (int kc = 0; kc < Kd; kc += 32) {
    int kb = kc + half * 8;   // A frag: lane L -> M=L%16, K in [kb,kb+8) U [kb+16,kb+24)
    v16h a = join8(*(const v8h*)(xrow + kb), *(const v8h*)(xrow + kb + 16));
    v16h b;  // B frag: lane L -> N=L%16, K same as A frag
    for (int s = 0; s < 16; ++s) b[s] = W[(size_t)(kc + (s < 8 ? half * 8 + s : 8 + half * 8 + s)) * N + nt * 16 + m];
    acc = __builtin_amdgcn_wmma_f32_16x16x32_f16(false, a, false, b, (short)0, acc,
                                                 false, false);
  }
  int ncol = nt * 16 + m;              // D: lane L -> N=L%16, M=r+8*(L/16)
  float bv = bias ? bias[ncol] : 0.f;
#pragma unroll
  for (int r = 0; r < 8; ++r) {
    float v = acc[r] + bv;
    if (relu) v = fmaxf(v, 0.f);
    out[(size_t)(mt * 16 + r + half * 8) * N + ncol] = v;
  }
}

// ---------------------------------------------------------------- LayerNorm
template <int NCOLS>
__global__ void k_ln(const float* __restrict__ x, const float* __restrict__ g,
                     const float* __restrict__ b, float* __restrict__ y, int M) {
  int row = blockIdx.x * (blockDim.x >> 5) + (threadIdx.x >> 5);
  if (row >= M) return;
  int lane = threadIdx.x & 31;
  const int PER = NCOLS / 32;
  const float* xr = x + (size_t)row * NCOLS;
  float vals[PER];
  float s = 0.f;
#pragma unroll
  for (int i = 0; i < PER; ++i) { vals[i] = xr[lane + i * 32]; s += vals[i]; }
#pragma unroll
  for (int o = 16; o > 0; o >>= 1) s += __shfl_xor(s, o, 32);
  float mean = s * (1.0f / NCOLS);
  float q = 0.f;
#pragma unroll
  for (int i = 0; i < PER; ++i) { float d = vals[i] - mean; q += d * d; }
#pragma unroll
  for (int o = 16; o > 0; o >>= 1) q += __shfl_xor(q, o, 32);
  float rstd = rsqrtf(q * (1.0f / NCOLS) + 1e-5f);
  float* yr = y + (size_t)row * NCOLS;
#pragma unroll
  for (int i = 0; i < PER; ++i) {
    int c = lane + i * 32;
    yr[c] = (vals[i] - mean) * rstd * g[c] + b[c];
  }
}

// ----------------------------------------------------------------- edge MLP
// h1[e,r]=relu(P[i,r]+Q[j,r]) (b1 folded into P); h2=relu(h1@W2+b2);
// A[i,j]=sigmoid(h2.w3+b3). One wave = 1 u-row x 16 v-cols.
__global__ __launch_bounds__(256)
void k_edge(const float* __restrict__ P, const float* __restrict__ Q,
            const _Float16* __restrict__ W2,  // [RDIM, KDIM] f16 row-major
            const float* __restrict__ b2, const float* __restrict__ w3,
            const float* __restrict__ b3, float* __restrict__ A) {
  __shared__ _Float16 w2s[RDIM * KDIM];  // 64 KB, XOR-swizzled in 8-dword blocks
  // Stage W2 with direct global->LDS async copies (no VGPR round-trip).
  // Swizzled LDS byte address per 16B chunk c (row = c>>4, cp4 = c&15, s = row&7):
  //   byte = row*256 + 32*((cp4>>1) ^ s) + 16*(cp4&1)
  // which equals the dword mapping row*64 + (cp ^ (s<<3)) used by the read side.
  {
    unsigned lbase = (unsigned)(size_t)(void*)w2s;  // low 32 bits == LDS address
#pragma unroll
    for (int t = 0; t < 16; ++t) {
      int c = (int)threadIdx.x + t * 256;           // 0..4095 16B chunks
      int row = c >> 4;
      int cp4 = c & 15;
      int s_ = row & 7;
      unsigned lds = lbase + (unsigned)(row * 256 + 32 * ((cp4 >> 1) ^ s_) +
                                        16 * (cp4 & 1));
      const char* g = (const char*)W2 + (size_t)c * 16;
      asm volatile("global_load_async_to_lds_b128 %0, %1, off"
                   :: "v"(lds), "v"(g) : "memory");
    }
    asm volatile("s_wait_asynccnt 0x0" ::: "memory");
  }
  __syncthreads();

  int wave = threadIdx.x >> 5;
  int lane = threadIdx.x & 31;
  int task = blockIdx.x * 8 + wave;  // 0..16383
  int i = task >> 5;                 // u row
  int j0 = (task & 31) << 4;         // v col base
  int m = lane & 15;
  int half = lane >> 4;
  const float* prow = P + (size_t)i * RDIM;
  const float* qrow = Q + (size_t)(j0 + m) * RDIM;

  float b2v[8], w3v[8];
#pragma unroll
  for (int nt = 0; nt < 8; ++nt) { int n = nt * 16 + m; b2v[nt] = b2[n]; w3v[nt] = w3[n]; }

  v8f zero = {};
  v8f acc[8];
#pragma unroll
  for (int nt = 0; nt < 8; ++nt) acc[nt] = zero;

#pragma unroll
  for (int kc = 0; kc < 8; ++kc) {
    int kb = kc * 32 + half * 8;
    v4f p0 = *(const v4f*)(prow + kb);
    v4f p1 = *(const v4f*)(prow + kb + 4);
    v4f p2 = *(const v4f*)(prow + kb + 16);
    v4f p3 = *(const v4f*)(prow + kb + 20);
    v4f q0 = *(const v4f*)(qrow + kb);
    v4f q1 = *(const v4f*)(qrow + kb + 4);
    v4f q2 = *(const v4f*)(qrow + kb + 16);
    v4f q3 = *(const v4f*)(qrow + kb + 20);
    v16h a;
#pragma unroll
    for (int t = 0; t < 4; ++t) {
      a[t]      = (_Float16)fmaxf(p0[t] + q0[t], 0.f);
      a[t + 4]  = (_Float16)fmaxf(p1[t] + q1[t], 0.f);
      a[t + 8]  = (_Float16)fmaxf(p2[t] + q2[t], 0.f);
      a[t + 12] = (_Float16)fmaxf(p3[t] + q3[t], 0.f);
    }
#pragma unroll
    for (int nt = 0; nt < 8; ++nt) {
      v16h bfr;
#pragma unroll
      for (int s = 0; s < 16; ++s) {
        int row = kc * 32 + (s < 8 ? half * 8 + s : 8 + half * 8 + s);
        bfr[s] = w2s[row * KDIM + ((nt ^ (row & 7)) << 4) + m];
      }
      acc[nt] = __builtin_amdgcn_wmma_f32_16x16x32_f16(false, a, false, bfr, (short)0,
                                                       acc[nt], false, false);
    }
  }

  float b3v = b3[0];
  float s[8];
#pragma unroll
  for (int r = 0; r < 8; ++r) {
    float t = 0.f;
#pragma unroll
    for (int nt = 0; nt < 8; ++nt) {
      float v = acc[nt][r] + b2v[nt];
      v = fmaxf(v, 0.f);
      t += v * w3v[nt];
    }
#pragma unroll
    for (int o = 8; o > 0; o >>= 1) t += __shfl_xor(t, o, 32);
    s[r] = t;
  }
  if (m == 0) {
    float* arow = A + (size_t)i * NV + j0 + half * 8;  // edge m_e = r + 8*half
#pragma unroll
    for (int r = 0; r < 8; ++r) {
      float x = s[r] + b3v;
      arow[r] = 1.0f / (1.0f + __expf(-x));
    }
  }
}

// ------------------------------------------------------------------- driver
extern "C" void kernel_launch(void* const* d_in, const int* in_sizes, int n_in,
                              void* d_out, int out_size, void* d_ws, size_t ws_size,
                              hipStream_t stream) {
  (void)in_sizes; (void)n_in; (void)out_size; (void)ws_size;
  const float* in_eu  = (const float*)d_in[0];
  const float* in_ev  = (const float*)d_in[1];
  const float* adj_w1 = (const float*)d_in[4];
  const float* adj_b1 = (const float*)d_in[5];
  const float* adj_w2 = (const float*)d_in[6];
  const float* adj_b2 = (const float*)d_in[7];
  const float* adj_w3 = (const float*)d_in[8];
  const float* adj_b3 = (const float*)d_in[9];
  const float* vtu_w  = (const float*)d_in[10];
  const float* vtu_b  = (const float*)d_in[11];
  const float* vtu_g  = (const float*)d_in[12];
  const float* vtu_bb = (const float*)d_in[13];
  const float* utv_w  = (const float*)d_in[14];
  const float* utv_b  = (const float*)d_in[15];
  const float* utv_g  = (const float*)d_in[16];
  const float* utv_bb = (const float*)d_in[17];
  const float* uupd_w = (const float*)d_in[18];
  const float* uupd_g = (const float*)d_in[19];
  const float* uupd_bb= (const float*)d_in[20];
  const float* vupd_w = (const float*)d_in[21];
  const float* vupd_g = (const float*)d_in[22];
  const float* vupd_bb= (const float*)d_in[23];

  char* ws = (char*)d_ws;
  size_t off = 0;
  auto alloc = [&](size_t bytes) -> char* {
    char* p = ws + off;
    off += (bytes + 255) & ~(size_t)255;
    return p;
  };
  float* eu   = (float*)alloc((size_t)NU * KDIM * 4);
  float* ev   = (float*)alloc((size_t)NV * KDIM * 4);
  float* A    = (float*)alloc((size_t)NU * NV * 4);
  float* P    = (float*)alloc((size_t)NU * RDIM * 4);
  float* Q    = (float*)alloc((size_t)NV * RDIM * 4);
  float* Mv   = (float*)alloc((size_t)NV * RDIM * 4);
  float* mpre = (float*)alloc((size_t)NU * RDIM * 4);
  float* msg  = (float*)alloc((size_t)NU * RDIM * 4);
  float* upre = (float*)alloc((size_t)NU * KDIM * 4);
  _Float16* eu16 = (_Float16*)alloc((size_t)NU * KDIM * 2);
  _Float16* ev16 = (_Float16*)alloc((size_t)NV * KDIM * 2);
  _Float16* A16  = (_Float16*)alloc((size_t)NU * NV * 2);
  _Float16* M16  = (_Float16*)alloc((size_t)NV * RDIM * 2);
  _Float16* Z16  = (_Float16*)alloc((size_t)NU * (KDIM + RDIM) * 2);
  _Float16* w1h  = (_Float16*)alloc((size_t)2 * KDIM * RDIM * 2);
  _Float16* w2h  = (_Float16*)alloc((size_t)RDIM * KDIM * 2);
  _Float16* wvtu = (_Float16*)alloc((size_t)KDIM * RDIM * 2);
  _Float16* wutv = (_Float16*)alloc((size_t)KDIM * RDIM * 2);
  _Float16* wuup = (_Float16*)alloc((size_t)(KDIM + RDIM) * KDIM * 2);
  _Float16* wvup = (_Float16*)alloc((size_t)(KDIM + RDIM) * KDIM * 2);

  auto cvt = [&](const float* x, _Float16* y, int n) {
    k_cvt_f16<<<(n + 255) / 256, 256, 0, stream>>>(x, y, n);
  };
  auto gemm = [&](const _Float16* X, const _Float16* W, const float* bias, float* out,
                  int M, int Kd, int N, int relu) {
    int tiles = (M >> 4) * (N >> 4);
    k_gemm_f16<<<(tiles + 3) / 4, 128, 0, stream>>>(X, W, bias, out, M, Kd, N, relu);
  };

  // weight conversions (deterministic each call)
  cvt(adj_w1, w1h, 2 * KDIM * RDIM);
  cvt(adj_w2, w2h, RDIM * KDIM);
  cvt(vtu_w, wvtu, KDIM * RDIM);
  cvt(utv_w, wutv, KDIM * RDIM);
  cvt(uupd_w, wuup, (KDIM + RDIM) * KDIM);
  cvt(vupd_w, wvup, (KDIM + RDIM) * KDIM);
  k_copy_f32<<<(NU * KDIM + 255) / 256, 256, 0, stream>>>(in_eu, eu, NU * KDIM);
  k_copy_f32<<<(NV * KDIM + 255) / 256, 256, 0, stream>>>(in_ev, ev, NV * KDIM);

  for (int it = 0; it < 2; ++it) {
    cvt(eu, eu16, NU * KDIM);
    cvt(ev, ev16, NV * KDIM);
    gemm(eu16, w1h, adj_b1, P, NU, KDIM, RDIM, 0);                 // P = eu@W1a + b1
    gemm(ev16, w1h + (size_t)KDIM * RDIM, nullptr, Q, NV, KDIM, RDIM, 0);  // Q = ev@W1b
    gemm(ev16, wvtu, vtu_b, Mv, NV, KDIM, RDIM, 1);                // Mv = relu(ev@vtu+b)
    k_edge<<<(NU * (NV / 16)) / 8, 256, 0, stream>>>(P, Q, w2h, adj_b2, adj_w3, adj_b3, A);
    // U update
    cvt(A, A16, NU * NV);
    cvt(Mv, M16, NV * RDIM);
    gemm(A16, M16, nullptr, mpre, NU, NV, RDIM, 0);                // A @ Mv
    k_ln<RDIM><<<NU / 8, 256, 0, stream>>>(mpre, vtu_g, vtu_bb, msg, NU);
    k_concat_cvt<<<(NU * (KDIM + RDIM) + 255) / 256, 256, 0, stream>>>(eu, msg, Z16, NU,
                                                                       KDIM, RDIM);
    gemm(Z16, wuup, nullptr, upre, NU, KDIM + RDIM, KDIM, 0);
    k_ln<KDIM><<<NU / 8, 256, 0, stream>>>(upre, uupd_g, uupd_bb, eu, NU);
    // V update (uses updated eu)
    cvt(eu, eu16, NU * KDIM);
    gemm(eu16, wutv, utv_b, Mv, NU, KDIM, RDIM, 1);                // Mu = relu(eu@utv+b)
    cvt(Mv, M16, NU * RDIM);
    k_transpose_cvt<<<(NU * NV + 255) / 256, 256, 0, stream>>>(A, A16, NU, NV);  // A^T
    gemm(A16, M16, nullptr, mpre, NV, NU, RDIM, 0);                // A^T @ Mu
    k_ln<RDIM><<<NV / 8, 256, 0, stream>>>(mpre, utv_g, utv_bb, msg, NV);
    k_concat_cvt<<<(NV * (KDIM + RDIM) + 255) / 256, 256, 0, stream>>>(ev, msg, Z16, NV,
                                                                       KDIM, RDIM);
    gemm(Z16, wvup, nullptr, upre, NV, KDIM + RDIM, KDIM, 0);
    k_ln<KDIM><<<NV / 8, 256, 0, stream>>>(upre, vupd_g, vupd_bb, ev, NV);
  }

  float* out = (float*)d_out;
  k_copy_f32<<<(NU * KDIM + 255) / 256, 256, 0, stream>>>(eu, out, NU * KDIM);
  k_copy_f32<<<(NV * KDIM + 255) / 256, 256, 0, stream>>>(ev, out + NU * KDIM, NV * KDIM);
  k_copy_f32<<<(NU * NV + 255) / 256, 256, 0, stream>>>(A, out + NU * KDIM + NV * KDIM,
                                                        NU * NV);
}
